// VanillaAttention_23381801960097
// MI455X (gfx1250) — matrix operand order repaired and checked by reading, not compiled, not run on hardware
//
#include <hip/hip_runtime.h>

typedef __bf16 bf16;
typedef __attribute__((ext_vector_type(16))) __bf16 v16bf;
typedef __attribute__((ext_vector_type(8)))  __bf16 v8bf;
typedef __attribute__((ext_vector_type(8)))  float  v8f;

#define B_SZ 8
#define N_SZ 1024
#define D_SZ 1024
#define H_SZ 16
#define HD   64
#define SCALE 0.125f
#define NEGV  (-1000000000.0f)

#if __has_builtin(__builtin_amdgcn_s_wait_asynccnt)
#define WAIT_ASYNC0() __builtin_amdgcn_s_wait_asynccnt(0)
#else
#define WAIT_ASYNC0() asm volatile("s_wait_asynccnt 0" ::: "memory")
#endif

__device__ __forceinline__ bf16 f2bf(float f) {
  unsigned u = __builtin_bit_cast(unsigned, f);
  unsigned r = u + 0x7FFFu + ((u >> 16) & 1u);   // round-to-nearest-even
  unsigned short h = (unsigned short)(r >> 16);
  return __builtin_bit_cast(bf16, h);
}

__device__ __forceinline__ v8f wmma_bf16(v16bf a, v16bf b, v8f c) {
  return __builtin_amdgcn_wmma_f32_16x16x32_bf16(
      false, a, false, b, (short)0, c, false, false);
}

union FragA { v16bf v; v8bf h[2]; };

// ---------------------------------------------------------------------------
// f32 -> bf16 elementwise convert
// ---------------------------------------------------------------------------
__global__ __launch_bounds__(256) void cvt_f32_bf16(
    const float* __restrict__ in, bf16* __restrict__ out, int n) {
  int i = blockIdx.x * 256 + threadIdx.x;
  if (i < n) out[i] = f2bf(in[i]);
}

// ---------------------------------------------------------------------------
// C[M,Nt] = A[M,Kd] @ W[Nt,Kd]^T   (torch Linear convention)
// mode 0: outB row-major bf16 [M,Nt]
// mode 1: outB = per-head-transposed V:  Vt[b][h][x][j]
// mode 2: outF f32 = (C + bias) * rowmask
// Block 256 thr = 8 waves (4m x 2n); block tile 128x128; wave tile 32x64.
// Weight tile (128 cols x 32 K, 8 KB) staged to LDS through the async-copy
// engine (GLOBAL_LOAD_ASYNC_TO_LDS_B128, ASYNCcnt), double-buffered.
// A fragments stream directly from global (private per block-row).
// All 4 B fragments are preloaded per k-step so the 8 ds_load_b128 can be
// claused and the 8 WMMAs issue back-to-back with partial dscnt waits.
// ---------------------------------------------------------------------------
__global__ __launch_bounds__(256) void gemm_bf16_wmma(
    const bf16* __restrict__ A, const bf16* __restrict__ W,
    int Kd, int Nt, int mode,
    const float* __restrict__ bias, const unsigned char* __restrict__ mask,
    bf16* __restrict__ outB, float* __restrict__ outF) {
  __shared__ __align__(64) bf16 Bs[2][128 * 32];   // 2 x 8 KB weight tiles

  const int t    = threadIdx.x;
  const int lane = t & 31;
  const int wave = t >> 5;
  const int half = lane >> 4;       // 0/1: which 16-lane group
  const int lr   = lane & 15;
  const int wm   = wave >> 1;       // 0..3
  const int wn   = wave & 1;        // 0..1
  const int m0   = blockIdx.y * 128 + wm * 32;
  const int n0   = blockIdx.x * 128;   // block column base
  const int nw0  = wn * 64;            // wave column base within block

  const bf16* arow[2];
  arow[0] = A + (size_t)(m0 + lr) * Kd;
  arow[1] = A + (size_t)(m0 + 16 + lr) * Kd;

  const unsigned ldsBase = (unsigned)(size_t)(void*)&Bs[0][0];

  // Each thread copies 2 x 16-byte segments per tile (512 segs = 8 KB).
  const int seg0 = t, seg1 = t + 256;
  const int r0 = seg0 >> 2, c0 = (seg0 & 3) * 8;   // col in elements
  const int r1 = seg1 >> 2, c1 = (seg1 & 3) * 8;

  auto issueTile = [&](int buf, int k0) {
    unsigned l0 = ldsBase + (unsigned)(buf * 8192 + r0 * 64 + c0 * 2);
    unsigned l1 = ldsBase + (unsigned)(buf * 8192 + r1 * 64 + c1 * 2);
    unsigned g0 = (unsigned)(((n0 + r0) * Kd + k0 + c0) * 2);
    unsigned g1 = (unsigned)(((n0 + r1) * Kd + k0 + c1) * 2);
    asm volatile("global_load_async_to_lds_b128 %0, %1, %2"
                 :: "v"(l0), "v"(g0), "s"(W) : "memory");
    asm volatile("global_load_async_to_lds_b128 %0, %1, %2"
                 :: "v"(l1), "v"(g1), "s"(W) : "memory");
  };

  v8f acc[2][4] = {};
  const int nk = Kd >> 5;

  issueTile(0, 0);
  for (int kk = 0; kk < nk; ++kk) {
    WAIT_ASYNC0();                      // my async writes for buf[kk&1] done
    __syncthreads();                    // everyone's writes visible
    if (kk + 1 < nk) issueTile((kk + 1) & 1, (kk + 1) * 32);

    const int k0 = kk * 32;
    FragA a[2];
#pragma unroll
    for (int mt = 0; mt < 2; ++mt) {
      // A 16x32 bf16 fragment: lane holds K {half*16..+15}
      a[mt].h[0] = *(const v8bf*)(arow[mt] + k0 + half * 16);
      a[mt].h[1] = *(const v8bf*)(arow[mt] + k0 + half * 16 + 8);
    }
    const bf16* bbase = &Bs[kk & 1][0];
    v16bf bfr[4];
#pragma unroll
    for (int nt = 0; nt < 4; ++nt) {
      // B 32x16 fragment: lane = column n, K = half*16 + e (contig in LDS row)
      int nrel = nw0 + nt * 16 + lr;
      bfr[nt] = *(const v16bf*)(bbase + nrel * 32 + half * 16);
    }
#pragma unroll
    for (int nt = 0; nt < 4; ++nt)
#pragma unroll
      for (int mt = 0; mt < 2; ++mt)
        acc[mt][nt] = wmma_bf16(a[mt].v, bfr[nt], acc[mt][nt]);
  }

#pragma unroll
  for (int mt = 0; mt < 2; ++mt) {
#pragma unroll
    for (int nt = 0; nt < 4; ++nt) {
#pragma unroll
      for (int r = 0; r < 8; ++r) {
        int m = m0 + mt * 16 + half * 8 + r;     // C/D: M = r + 8*half
        int n = n0 + nw0 + nt * 16 + lr;         // N = lane%16
        float v = acc[mt][nt][r];
        if (mode == 0) {
          outB[(size_t)m * Nt + n] = f2bf(v);
        } else if (mode == 1) {
          int b = m >> 10, j = m & (N_SZ - 1);
          int h = n >> 6,  x = n & (HD - 1);
          outB[(((size_t)b * H_SZ + h) * HD + x) * N_SZ + j] = f2bf(v);
        } else {
          outF[(size_t)m * Nt + n] = mask[m] ? (v + bias[n]) : 0.0f;
        }
      }
    }
  }
}

// ---------------------------------------------------------------------------
// Attention: scores = keys @ queries^T * 0.125 (+mask), softmax over j,
//            O = P @ V.   One block per (b, h, 16-row i-tile), 256 threads.
// LDS: 64 KB score buffer, reused for bf16 P (32 KB) + f32 partial O (4 KB).
// ---------------------------------------------------------------------------
__global__ __launch_bounds__(256) void attn_wmma(
    const bf16* __restrict__ Kb,   // keys   [B,N,D] bf16
    const bf16* __restrict__ Qb,   // queries[B,N,D] bf16
    const bf16* __restrict__ Vt,   // values transposed [B,H,HD,N] bf16
    const unsigned char* __restrict__ mask,
    bf16* __restrict__ Ob) {       // out [B,N,D] bf16
  __shared__ __align__(64) float Smem[16 * N_SZ];            // 64 KB
  bf16*  Pmem = (bf16*)Smem;                                 // first 32 KB
  float* Of   = (float*)((char*)Smem + 32768);               // 4 KB (16x64)

  const int t    = threadIdx.x;
  const int lane = t & 31;
  const int wave = t >> 5;
  const int half = lane >> 4;
  const int lr   = lane & 15;
  const int i0   = blockIdx.x * 16;
  const int h    = blockIdx.y;
  const int b    = blockIdx.z;

  // --- phase 1: S = keys @ queries^T, scaled + masked, into LDS (f32) ---
  unsigned char mi[8];
#pragma unroll
  for (int r = 0; r < 8; ++r)
    mi[r] = mask[b * N_SZ + i0 + half * 8 + r];

  const bf16* kptr = Kb + ((size_t)(b * N_SZ) + i0 + lr) * D_SZ + h * HD;
  FragA aK[2];
#pragma unroll
  for (int ks = 0; ks < 2; ++ks) {
    aK[ks].h[0] = *(const v8bf*)(kptr + ks * 32 + half * 16);
    aK[ks].h[1] = *(const v8bf*)(kptr + ks * 32 + half * 16 + 8);
  }

  for (int q = 0; q < 8; ++q) {              // 8 j-tiles per wave
    int j0 = (wave * 8 + q) * 16;
    int jc = j0 + lr;
    unsigned char mj = mask[b * N_SZ + jc];
    const bf16* qptr = Qb + ((size_t)(b * N_SZ) + jc) * D_SZ + h * HD;
    v8f acc = {};
#pragma unroll
    for (int ks = 0; ks < 2; ++ks) {
      v16bf bq = *(const v16bf*)(qptr + ks * 32 + 16 * half);
      acc = wmma_bf16(aK[ks].v, bq, acc);
    }
#pragma unroll
    for (int r = 0; r < 8; ++r) {
      int m = half * 8 + r;
      float s = acc[r] * SCALE + ((mi[r] && mj) ? 0.0f : NEGV);
      Smem[m * N_SZ + jc] = s;
    }
  }
  __syncthreads();

  // --- phase 2: softmax over j (16 threads per row, 64 elems each) ---
  {
    int row = t >> 4;
    int c16 = t & 15;
    int base = row * N_SZ + c16 * 64;
    float vbuf[64];
    float mx = -3.4e38f;
#pragma unroll
    for (int k = 0; k < 64; ++k) {
      vbuf[k] = Smem[base + k];
      mx = fmaxf(mx, vbuf[k]);
    }
#pragma unroll
    for (int off = 1; off < 16; off <<= 1)
      mx = fmaxf(mx, __shfl_xor(mx, off, 32));
    float sum = 0.0f;
#pragma unroll
    for (int k = 0; k < 64; ++k) {
      vbuf[k] = __expf(vbuf[k] - mx);
      sum += vbuf[k];
    }
#pragma unroll
    for (int off = 1; off < 16; off <<= 1)
      sum += __shfl_xor(sum, off, 32);
    float inv = 1.0f / sum;
    __syncthreads();                         // all S reads done before P overwrite
#pragma unroll
    for (int k = 0; k < 64; ++k)
      Pmem[base + k] = f2bf(vbuf[k] * inv);
#pragma unroll
    for (int z = 0; z < 4; ++z)
      Of[t + z * 256] = 0.0f;
  }
  __syncthreads();

  // --- phase 3: O = P @ V ; each wave handles a 128-wide j slab ---
  v8f accO[4] = {};
  int jb = wave * 128;
  for (int kk = 0; kk < 4; ++kk) {
    int k0 = jb + kk * 32;
    FragA aP;
    aP.h[0] = *(const v8bf*)(Pmem + lr * N_SZ + k0 + half * 16);
    aP.h[1] = *(const v8bf*)(Pmem + lr * N_SZ + k0 + half * 16 + 8);
#pragma unroll
    for (int nt = 0; nt < 4; ++nt) {
      const bf16* vptr = Vt + (((size_t)b * H_SZ + h) * HD + nt * 16 + lr) * N_SZ;
      v16bf bv = *(const v16bf*)(vptr + k0 + 16 * half);
      accO[nt] = wmma_bf16(aP.v, bv, accO[nt]);
    }
  }
#pragma unroll
  for (int nt = 0; nt < 4; ++nt)
#pragma unroll
    for (int r = 0; r < 8; ++r)
      atomicAdd(&Of[(half * 8 + r) * HD + nt * 16 + lr], accO[nt][r]);
  __syncthreads();

  for (int idx = t; idx < 16 * HD; idx += 256) {
    int m = idx >> 6, x = idx & (HD - 1);
    Ob[((size_t)(b * N_SZ) + i0 + m) * D_SZ + h * HD + x] = f2bf(Of[idx]);
  }
}

// ---------------------------------------------------------------------------
extern "C" void kernel_launch(void* const* d_in, const int* in_sizes, int n_in,
                              void* d_out, int out_size, void* d_ws, size_t ws_size,
                              hipStream_t stream) {
  (void)in_sizes; (void)n_in; (void)out_size; (void)ws_size;
  const float* x  = (const float*)d_in[0];
  const unsigned char* mask = (const unsigned char*)d_in[1];  // jnp bool -> 1 byte
  const float* Wk = (const float*)d_in[2];
  const float* Wq = (const float*)d_in[3];
  const float* Wv = (const float*)d_in[4];
  const float* Wo = (const float*)d_in[5];
  const float* bo = (const float*)d_in[6];
  float* out = (float*)d_out;

  const size_t MB = 1u << 20;
  char* ws = (char*)d_ws;
  bf16* xb   = (bf16*)(ws + 0 * MB);    // 16 MB
  bf16* Wkb  = (bf16*)(ws + 16 * MB);   //  2 MB
  bf16* Wqb  = (bf16*)(ws + 18 * MB);   //  2 MB
  bf16* Wvb  = (bf16*)(ws + 20 * MB);   //  2 MB
  bf16* Wob  = (bf16*)(ws + 22 * MB);   //  2 MB
  bf16* Qbuf = (bf16*)(ws + 24 * MB);   // 16 MB queries
  bf16* Kbuf = (bf16*)(ws + 40 * MB);   // 16 MB keys
  bf16* Vt   = (bf16*)(ws + 56 * MB);   // 16 MB values (per-head transposed)
  bf16* Obuf = (bf16*)(ws + 72 * MB);   // 16 MB attention output

  const int nX = B_SZ * N_SZ * D_SZ;    // 8388608
  const int nW = D_SZ * D_SZ;           // 1048576

  cvt_f32_bf16<<<(nX + 255) / 256, 256, 0, stream>>>(x,  xb,  nX);
  cvt_f32_bf16<<<(nW + 255) / 256, 256, 0, stream>>>(Wk, Wkb, nW);
  cvt_f32_bf16<<<(nW + 255) / 256, 256, 0, stream>>>(Wq, Wqb, nW);
  cvt_f32_bf16<<<(nW + 255) / 256, 256, 0, stream>>>(Wv, Wvb, nW);
  cvt_f32_bf16<<<(nW + 255) / 256, 256, 0, stream>>>(Wo, Wob, nW);

  dim3 ggrid(D_SZ / 128, (B_SZ * N_SZ) / 128);  // (8, 64)
  // queries = x @ Wq^T ; keys = x @ Wk^T ; values = x @ Wv^T (transposed layout)
  gemm_bf16_wmma<<<ggrid, 256, 0, stream>>>(xb, Wqb, D_SZ, D_SZ, 0,
                                            nullptr, nullptr, Qbuf, nullptr);
  gemm_bf16_wmma<<<ggrid, 256, 0, stream>>>(xb, Wkb, D_SZ, D_SZ, 0,
                                            nullptr, nullptr, Kbuf, nullptr);
  gemm_bf16_wmma<<<ggrid, 256, 0, stream>>>(xb, Wvb, D_SZ, D_SZ, 1,
                                            nullptr, nullptr, Vt, nullptr);

  dim3 agrid(N_SZ / 16, H_SZ, B_SZ);            // (64, 16, 8)
  attn_wmma<<<agrid, 256, 0, stream>>>(Kbuf, Qbuf, Vt, mask, Obuf);

  // out = Obuf @ Wo^T + bo, zeroed at padded rows
  gemm_bf16_wmma<<<ggrid, 256, 0, stream>>>(Obuf, Wob, D_SZ, D_SZ, 2,
                                            bo, mask, nullptr, out);
}
